// TransformerLayer_41162966565467
// MI455X (gfx1250) — matrix operand order repaired and checked by reading, not compiled, not run on hardware
//
#include <hip/hip_runtime.h>

typedef unsigned short u16;
typedef unsigned int   u32;
typedef __attribute__((ext_vector_type(16))) __bf16 v16bf;
typedef __attribute__((ext_vector_type(8)))  float  v8f;

union Frag16 { v16bf v; uint4 q[2]; };

__device__ __forceinline__ u16 f2bf(float f) {
    union { float f; u32 u; } c; c.f = f;
    u32 u = c.u;
    u32 r = (u + 0x7FFFu + ((u >> 16) & 1u)) >> 16;   // round-to-nearest-even
    return (u16)r;
}

static constexpr int E_  = 1024;
static constexpr int H_  = 16;
static constexpr int R_  = 512;
static constexpr int LL  = 2048;
static constexpr int BB  = 2;
static constexpr int MR  = BB * LL;     // 4096 token rows
static constexpr int H3  = 3 * E_;      // 3072

// ---------------------------------------------------------------- fp32 -> bf16
__global__ __launch_bounds__(256) void cvt_kernel(const float* __restrict__ in,
                                                  u16* __restrict__ out, int n) {
    int i = blockIdx.x * 256 + threadIdx.x;
    if (i < n) out[i] = f2bf(in[i]);
}

// ---------------------------------------------------------------- LayerNorm (fp32 in, bf16 out)
__global__ __launch_bounds__(256) void ln_kernel(const float* __restrict__ x,
                                                 const float* __restrict__ g,
                                                 const float* __restrict__ b,
                                                 u16* __restrict__ out) {
    int row = blockIdx.x;
    int tid = threadIdx.x;
    int lane = tid & 31, wv = tid >> 5;
    const float4 v = ((const float4*)(x + (size_t)row * E_))[tid];
    float s  = v.x + v.y + v.z + v.w;
    float s2 = v.x * v.x + v.y * v.y + v.z * v.z + v.w * v.w;
#pragma unroll
    for (int m = 16; m >= 1; m >>= 1) { s += __shfl_xor(s, m); s2 += __shfl_xor(s2, m); }
    __shared__ float rs[8], rs2[8];
    if (lane == 0) { rs[wv] = s; rs2[wv] = s2; }
    __syncthreads();
    float ts = 0.f, ts2 = 0.f;
#pragma unroll
    for (int i = 0; i < 8; ++i) { ts += rs[i]; ts2 += rs2[i]; }
    float mu  = ts * (1.0f / E_);
    float var = ts2 * (1.0f / E_) - mu * mu;
    float inv = rsqrtf(var + 1e-5f);
    const float4 gg = ((const float4*)g)[tid];
    const float4 bb = ((const float4*)b)[tid];
    u16* orow = out + (size_t)row * E_ + tid * 4;
    orow[0] = f2bf((v.x - mu) * inv * gg.x + bb.x);
    orow[1] = f2bf((v.y - mu) * inv * gg.y + bb.y);
    orow[2] = f2bf((v.z - mu) * inv * gg.z + bb.z);
    orow[3] = f2bf((v.w - mu) * inv * gg.w + bb.w);
}

// ---------------------------------------------------------------- WMMA GEMM
//  C[M,N] = A[M,K](bf16) @ W[N,K](bf16)^T + bias, optional relu/resid/qscale
//  block tile 128x128, BK=32, 8 waves as 2(M) x 4(N), wave tile 64x32.
//  Global->LDS staging uses double-buffered GLOBAL_LOAD_ASYNC_TO_LDS_B128
//  (ASYNCcnt) so the copy of stage k+1 overlaps the WMMAs of stage k.
template<bool RELU, bool QSCALE, bool HAS_RES, bool OUTF, bool OUTB>
__global__ __launch_bounds__(256) void gemm_kernel(const u16* __restrict__ A,
                                                   const u16* __restrict__ W,
                                                   const float* __restrict__ bias,
                                                   const float* __restrict__ resid,
                                                   float* __restrict__ outF,
                                                   u16* __restrict__ outB,
                                                   int M, int N, int K) {
    __shared__ __align__(16) u16 sA[2][128 * 32];
    __shared__ __align__(16) u16 sB[2][128 * 32];
    const int tid  = threadIdx.x;
    const int lane = tid & 31;
    const int lo   = lane & 15, hi = lane >> 4;
    const int wv   = tid >> 5;
    const int wr   = wv >> 2;          // 0..1  (M)
    const int wc   = wv & 3;           // 0..3  (N)
    const int bm   = blockIdx.y * 128;
    const int bn   = blockIdx.x * 128;

    v8f acc[4][2] = {};

    const int ldr  = tid >> 1;           // 0..127
    const int lds_ = (tid & 1) * 16;     // 0 / 16

    const u16* gA = A + (size_t)(bm + ldr) * K + lds_;
    const u16* gW = W + (size_t)(bn + ldr) * K + lds_;

    // async-copy one 32-wide K stage into LDS buffer `buf`
    auto issue_stage = [&](int k0, int buf) {
        const u32 la = (u32)(uintptr_t)(&sA[buf][ldr * 32 + lds_]);   // LDS byte offset (addr[31:0])
        const u32 lb = (u32)(uintptr_t)(&sB[buf][ldr * 32 + lds_]);
        const u16* pa = gA + k0;
        const u16* pw = gW + k0;
        // INST_OFFSET applies to both LDS and memory sides (ISA 08 §4.4)
        asm volatile("global_load_async_to_lds_b128 %0, %1, off"           :: "v"(la), "v"(pa) : "memory");
        asm volatile("global_load_async_to_lds_b128 %0, %1, off offset:16" :: "v"(la), "v"(pa) : "memory");
        asm volatile("global_load_async_to_lds_b128 %0, %1, off"           :: "v"(lb), "v"(pw) : "memory");
        asm volatile("global_load_async_to_lds_b128 %0, %1, off offset:16" :: "v"(lb), "v"(pw) : "memory");
    };

    issue_stage(0, 0);                   // prime the pipeline (ASYNCcnt = 4)

    for (int k0 = 0; k0 < K; k0 += 32) {
        const int cur = (k0 >> 5) & 1;
        if (k0 + 32 < K) {
            issue_stage(k0 + 32, cur ^ 1);                       // cnt -> 8
            asm volatile("s_wait_asynccnt 0x4" ::: "memory");    // stage `cur` landed
        } else {
            asm volatile("s_wait_asynccnt 0x0" ::: "memory");
        }
        __syncthreads();                 // publish everyone's async LDS writes

        Frag16 af[4], bf[2];
#pragma unroll
        for (int mi = 0; mi < 4; ++mi) {
            int m  = wr * 64 + mi * 16 + lo;
            int kb = hi * 8;                       // A-frag: K = kb..kb+7, kb+16..kb+23
            af[mi].q[0] = *(const uint4*)(&sA[cur][m * 32 + kb]);
            af[mi].q[1] = *(const uint4*)(&sA[cur][m * 32 + kb + 16]);
        }
#pragma unroll
        for (int ni = 0; ni < 2; ++ni) {
            int n  = wc * 32 + ni * 16 + lo;
            int kb = hi * 8;                       // B-frag: K = kb..kb+7, kb+16..kb+23
            bf[ni].q[0] = *(const uint4*)(&sB[cur][n * 32 + kb]);
            bf[ni].q[1] = *(const uint4*)(&sB[cur][n * 32 + kb + 16]);
        }
#pragma unroll
        for (int mi = 0; mi < 4; ++mi)
#pragma unroll
            for (int ni = 0; ni < 2; ++ni)
                acc[mi][ni] = __builtin_amdgcn_wmma_f32_16x16x32_bf16(
                    false, af[mi].v, false, bf[ni].v, (short)0, acc[mi][ni], false, false);
        __syncthreads();                 // done reading `cur`; next iter may overwrite it
    }

    // epilogue (C layout: VGPR r -> rows r / r+8; lane&15 -> col)
#pragma unroll
    for (int mi = 0; mi < 4; ++mi)
#pragma unroll
        for (int ni = 0; ni < 2; ++ni) {
            const int col = bn + wc * 32 + ni * 16 + lo;
            const float bv = bias[col];
#pragma unroll
            for (int r = 0; r < 8; ++r) {
                const int row = bm + wr * 64 + mi * 16 + r + hi * 8;
                float v = acc[mi][ni][r] + bv;
                if (RELU)   v = fmaxf(v, 0.0f);
                if (QSCALE) { if (col < E_) v *= 0.125f; }   // 1/sqrt(64) folded into Q
                if (HAS_RES) v += resid[(size_t)row * N + col];
                if (OUTF) outF[(size_t)row * N + col] = v;
                if (OUTB) outB[(size_t)row * N + col] = f2bf(v);
            }
        }
}

// ---------------------------------------------------------------- banded flash attention
//  grid (L/64, H, B), 128 threads = 4 waves, one 16-query tile per wave.
__global__ __launch_bounds__(128) void attn_kernel(const u16* __restrict__ qkv,
                                                   const float* __restrict__ res_pos,
                                                   u16* __restrict__ o) {
    const int lane = threadIdx.x & 31;
    const int wv   = threadIdx.x >> 5;
    const int lo   = lane & 15, hi = lane >> 4;
    const int h    = blockIdx.y;
    const int b    = blockIdx.z;
    const int qbase = (blockIdx.x * 4 + wv) * 16;
    const int qend  = qbase + 15;

    __shared__ __align__(16) u16 lds[4 * (16 * 32 + 64 * 32)];
    u16* pst = &lds[wv * (16 * 32 + 64 * 32)];   // P tile 16x32
    u16* vt  = pst + 16 * 32;                    // V^T tile 64x32

    const u16* base = qkv + (size_t)b * LL * H3;

    // Q fragments (pre-scaled by 1/8 in the QKV GEMM)
    Frag16 qf[2];
    {
        const u16* qp = base + (size_t)(qbase + lo) * H3 + h * 64;
        const int kb = hi * 8;
#pragma unroll
        for (int i = 0; i < 2; ++i) {
            qf[i].q[0] = *(const uint4*)(qp + i * 32 + kb);
            qf[i].q[1] = *(const uint4*)(qp + i * 32 + kb + 16);
        }
    }

    float m_run[8], l_run[8];
#pragma unroll
    for (int r = 0; r < 8; ++r) { m_run[r] = -3.0e38f; l_run[r] = 0.0f; }
    v8f oacc[4] = {};

    int klo = qbase - 512; if (klo < 0) klo = 0;

    for (int kb0 = klo; kb0 <= qend; kb0 += 32) {
        float s[2][8];
#pragma unroll
        for (int t = 0; t < 2; ++t) {
            const int kt = kb0 + t * 16;
            v8f sacc = {};
            if (kt <= qend) {                    // wave-uniform branch
                int key = kt + lo; if (key > LL - 1) key = LL - 1;
                const u16* kp = base + (size_t)key * H3 + E_ + h * 64;
                const int kb = hi * 8;
                Frag16 kf[2];
#pragma unroll
                for (int i = 0; i < 2; ++i) {    // B-frag straight from global (K is [key][d])
                    kf[i].q[0] = *(const uint4*)(kp + i * 32 + kb);
                    kf[i].q[1] = *(const uint4*)(kp + i * 32 + kb + 16);
                }
                sacc = __builtin_amdgcn_wmma_f32_16x16x32_bf16(
                    false, qf[0].v, false, kf[0].v, (short)0, sacc, false, false);
                sacc = __builtin_amdgcn_wmma_f32_16x16x32_bf16(
                    false, qf[1].v, false, kf[1].v, (short)0, sacc, false, false);
            }
#pragma unroll
            for (int r = 0; r < 8; ++r) {
                const int row = qbase + r + hi * 8;
                const int key = kt + lo;
                const int rel = row - key;
                const bool valid = (kt <= qend) && rel >= 0 && rel < R_;
                s[t][r] = valid ? (sacc[r] + res_pos[h * R_ + rel]) : -3.0e38f;
            }
        }

        // online softmax (rows live in 16-lane half-groups)
        float alpha[8];
#pragma unroll
        for (int r = 0; r < 8; ++r) {
            float tm = fmaxf(s[0][r], s[1][r]);
#pragma unroll
            for (int m = 8; m >= 1; m >>= 1) tm = fmaxf(tm, __shfl_xor(tm, m));
            const float mn = fmaxf(m_run[r], tm);
            alpha[r] = __expf(m_run[r] - mn);
            float p0 = __expf(s[0][r] - mn); if (s[0][r] < -1.0e30f) p0 = 0.0f;
            float p1 = __expf(s[1][r] - mn); if (s[1][r] < -1.0e30f) p1 = 0.0f;
            s[0][r] = p0; s[1][r] = p1;
            float rsum = p0 + p1;
#pragma unroll
            for (int m = 8; m >= 1; m >>= 1) rsum += __shfl_xor(rsum, m);
            l_run[r] = l_run[r] * alpha[r] + rsum;
            m_run[r] = mn;
        }
#pragma unroll
        for (int ni = 0; ni < 4; ++ni)
#pragma unroll
            for (int r = 0; r < 8; ++r) oacc[ni][r] *= alpha[r];

        // P (C layout) -> LDS bf16
#pragma unroll
        for (int r = 0; r < 8; ++r) {
            pst[(r + hi * 8) * 32 + lo]      = f2bf(s[0][r]);
            pst[(r + hi * 8) * 32 + 16 + lo] = f2bf(s[1][r]);
        }
        // V tile (32 keys x 64 dims) -> LDS transposed [dim][key]
        {
            int key = kb0 + lane; if (key > LL - 1) key = LL - 1;
            const u16* vp = base + (size_t)key * H3 + 2 * E_ + h * 64;
#pragma unroll
            for (int c = 0; c < 8; ++c) {
                const uint4 d = *(const uint4*)(vp + c * 8);
                const u16* u = (const u16*)&d;
#pragma unroll
                for (int j = 0; j < 8; ++j) vt[(c * 8 + j) * 32 + lane] = u[j];
            }
        }
        asm volatile("s_wait_dscnt 0" ::: "memory");   // in-wave LDS RAW guard

        // P as A-frag, V^T columns as B-frags
        Frag16 pf;
        {
            const int kb = hi * 8;
            pf.q[0] = *(const uint4*)(pst + lo * 32 + kb);
            pf.q[1] = *(const uint4*)(pst + lo * 32 + kb + 16);
        }
#pragma unroll
        for (int ni = 0; ni < 4; ++ni) {
            Frag16 vf;
            const int dim = ni * 16 + lo;
            vf.q[0] = *(const uint4*)(vt + dim * 32 + hi * 8);
            vf.q[1] = *(const uint4*)(vt + dim * 32 + hi * 8 + 16);
            oacc[ni] = __builtin_amdgcn_wmma_f32_16x16x32_bf16(
                false, pf.v, false, vf.v, (short)0, oacc[ni], false, false);
        }
    }

    // normalize + store o[B,L,E] bf16 (head-major columns)
#pragma unroll
    for (int ni = 0; ni < 4; ++ni)
#pragma unroll
        for (int r = 0; r < 8; ++r) {
            const int row = qbase + r + hi * 8;
            const float v = oacc[ni][r] * (1.0f / l_run[r]);
            o[(size_t)(b * LL + row) * E_ + h * 64 + ni * 16 + lo] = f2bf(v);
        }
}

// ---------------------------------------------------------------- launch
extern "C" void kernel_launch(void* const* d_in, const int* in_sizes, int n_in,
                              void* d_out, int out_size, void* d_ws, size_t ws_size,
                              hipStream_t stream) {
    const float* x         = (const float*)d_in[0];
    const float* res_pos   = (const float*)d_in[1];
    const float* in_proj_w = (const float*)d_in[2];
    const float* in_proj_b = (const float*)d_in[3];
    const float* out_w     = (const float*)d_in[4];
    const float* out_b     = (const float*)d_in[5];
    const float* w1        = (const float*)d_in[6];
    const float* b1        = (const float*)d_in[7];
    const float* w2        = (const float*)d_in[8];
    const float* b2        = (const float*)d_in[9];
    const float* ln1_g     = (const float*)d_in[10];
    const float* ln1_b     = (const float*)d_in[11];
    const float* ln2_g     = (const float*)d_in[12];
    const float* ln2_b     = (const float*)d_in[13];
    float* out = (float*)d_out;

    char* w = (char*)d_ws;
    u16* wqkv  = (u16*)w;  w += (size_t)3 * E_ * E_ * 2;
    u16* wout  = (u16*)w;  w += (size_t)E_ * E_ * 2;
    u16* w1b   = (u16*)w;  w += (size_t)4 * E_ * E_ * 2;
    u16* w2b   = (u16*)w;  w += (size_t)4 * E_ * E_ * 2;
    u16* xnorm = (u16*)w;  w += (size_t)MR * E_ * 2;      // reused as hnorm
    u16* qkv   = (u16*)w;  w += (size_t)MR * H3 * 2;
    u16* obf   = (u16*)w;  w += (size_t)MR * E_ * 2;
    float* x2  = (float*)w; w += (size_t)MR * E_ * 4;
    u16* h1    = (u16*)w;  w += (size_t)MR * 4 * E_ * 2;

    // weights -> bf16 (deterministic every call)
    {
        int n;
        n = 3 * E_ * E_; cvt_kernel<<<(n + 255) / 256, 256, 0, stream>>>(in_proj_w, wqkv, n);
        n = E_ * E_;     cvt_kernel<<<(n + 255) / 256, 256, 0, stream>>>(out_w, wout, n);
        n = 4 * E_ * E_; cvt_kernel<<<(n + 255) / 256, 256, 0, stream>>>(w1, w1b, n);
        n = 4 * E_ * E_; cvt_kernel<<<(n + 255) / 256, 256, 0, stream>>>(w2, w2b, n);
    }

    // LN1
    ln_kernel<<<MR, 256, 0, stream>>>(x, ln1_g, ln1_b, xnorm);

    // QKV = xnorm @ in_proj_w^T + b (Q pre-scaled by 1/8), bf16 out
    gemm_kernel<false, true, false, false, true>
        <<<dim3(H3 / 128, MR / 128), 256, 0, stream>>>(
            xnorm, wqkv, in_proj_b, nullptr, nullptr, qkv, MR, H3, E_);

    // banded attention
    attn_kernel<<<dim3(LL / 64, H_, BB), 128, 0, stream>>>(qkv, res_pos, obf);

    // x2 = x + o @ out_w^T + out_b   (fp32)
    gemm_kernel<false, false, true, true, false>
        <<<dim3(E_ / 128, MR / 128), 256, 0, stream>>>(
            obf, wout, out_b, x, x2, nullptr, MR, E_, E_);

    // LN2
    ln_kernel<<<MR, 256, 0, stream>>>(x2, ln2_g, ln2_b, xnorm);

    // h1 = relu(hnorm @ w1^T + b1)   bf16
    gemm_kernel<true, false, false, false, true>
        <<<dim3(4 * E_ / 128, MR / 128), 256, 0, stream>>>(
            xnorm, w1b, b1, nullptr, nullptr, h1, MR, 4 * E_, E_);

    // out = x2 + h1 @ w2^T + b2      fp32
    gemm_kernel<false, false, true, true, false>
        <<<dim3(E_ / 128, MR / 128), 256, 0, stream>>>(
            h1, w2b, b2, x2, out, nullptr, MR, E_, 4 * E_);
}
